// pc_tokenizer_6064493822431
// MI455X (gfx1250) — matrix operand order repaired and checked by reading, not compiled, not run on hardware
//
#include <hip/hip_runtime.h>
#include <hip/hip_bf16.h>

typedef __attribute__((ext_vector_type(16))) _Float16 v16h;
typedef __attribute__((ext_vector_type(8)))  _Float16 v8h;
typedef __attribute__((ext_vector_type(8)))  float    v8f;

#define K_NBR 32

// ----------------------------------------------------------------------------
// Furthest point sampling: one block per batch; dist[] lives in LDS.
// Matches jnp.argmax tie-break (lowest index), CUDA-compatible start at 0.
// ----------------------------------------------------------------------------
__global__ void fps_kernel(const float* __restrict__ p, int N, int npoint,
                           int* __restrict__ idx) {
    int b = blockIdx.x;
    const float* pb = p + (size_t)b * N * 3;
    int* ib = idx + (size_t)b * npoint;
    __shared__ float dist[8192];
    __shared__ float rv[256];
    __shared__ int   ri[256];
    __shared__ float lp[3];
    int tid = threadIdx.x;
    const int bs = 256;

    float x0 = pb[0], y0 = pb[1], z0 = pb[2];
    for (int i = tid; i < N; i += bs) {
        float dx = pb[i * 3] - x0, dy = pb[i * 3 + 1] - y0, dz = pb[i * 3 + 2] - z0;
        dist[i] = dx * dx + dy * dy + dz * dz;
    }
    if (tid == 0) ib[0] = 0;
    __syncthreads();

    for (int s = 1; s < npoint; s++) {
        float bv = -1.0f; int bi = 0;
        for (int i = tid; i < N; i += bs)
            if (dist[i] > bv) { bv = dist[i]; bi = i; }
        rv[tid] = bv; ri[tid] = bi;
        __syncthreads();
        for (int off = bs >> 1; off > 0; off >>= 1) {
            if (tid < off) {
                float ov = rv[tid + off]; int oi = ri[tid + off];
                if (ov > rv[tid] || (ov == rv[tid] && oi < ri[tid])) {
                    rv[tid] = ov; ri[tid] = oi;
                }
            }
            __syncthreads();
        }
        if (tid == 0) {
            int j = ri[0];
            ib[s] = j;
            lp[0] = pb[j * 3]; lp[1] = pb[j * 3 + 1]; lp[2] = pb[j * 3 + 2];
        }
        __syncthreads();
        float lx = lp[0], ly = lp[1], lz = lp[2];
        for (int i = tid; i < N; i += bs) {
            float dx = pb[i * 3] - lx, dy = pb[i * 3 + 1] - ly, dz = pb[i * 3 + 2] - lz;
            float d = dx * dx + dy * dy + dz * dz;
            if (d < dist[i]) dist[i] = d;
        }
        __syncthreads();
    }
}

// out (B,S,3) = p[b, idx[b,s], :]
__global__ void gather_p_kernel(const float* __restrict__ p, const int* __restrict__ idx,
                                float* __restrict__ out, int B, int S, int N) {
    int t = blockIdx.x * blockDim.x + threadIdx.x;
    if (t >= B * S) return;
    int b = t / S, s = t % S;
    int j = idx[b * S + s];
    const float* src = p + ((size_t)b * N + j) * 3;
    float* dst = out + ((size_t)b * S + s) * 3;
    dst[0] = src[0]; dst[1] = src[1]; dst[2] = src[2];
}

// out (B,C,S) f32 = f[b,c,idx[b,s]]   (f is channel-major (B,C,N))
__global__ void gather_f32_kernel(const float* __restrict__ f, const int* __restrict__ idx,
                                  float* __restrict__ out, int B, int C, int S, int N) {
    int t = blockIdx.x * blockDim.x + threadIdx.x;
    if (t >= B * C * S) return;
    int b = t / (C * S); int r = t % (C * S);
    int c = r / S, s = r % S;
    out[((size_t)b * C + c) * S + s] = f[((size_t)b * C + c) * N + idx[b * S + s]];
}

// out (B,S,C) f16 = f[b, idx[b,s], c]   (f is (B,N,C) row-major)
__global__ void gather_f16_kernel(const _Float16* __restrict__ f, const int* __restrict__ idx,
                                  _Float16* __restrict__ out, int B, int C, int S, int N) {
    int t = blockIdx.x * blockDim.x + threadIdx.x;
    if (t >= B * S * C) return;
    int b = t / (S * C); int r = t % (S * C);
    int s = r / C, c = r % C;
    out[t] = f[((size_t)b * N + idx[b * S + s]) * C + c];
}

// ----------------------------------------------------------------------------
// Ball query, CUDA semantics: first K hits by ascending index, pad with first.
// ----------------------------------------------------------------------------
__global__ void ball_query_kernel(const float* __restrict__ ctr, const float* __restrict__ pts,
                                  int* __restrict__ nbr, int B, int S, int N, float r2) {
    int t = blockIdx.x * blockDim.x + threadIdx.x;
    if (t >= B * S) return;
    int b = t / S, s = t % S;
    const float* c  = ctr + ((size_t)b * S + s) * 3;
    const float* pb = pts + (size_t)b * N * 3;
    int* nb = nbr + ((size_t)b * S + s) * K_NBR;
    float cx = c[0], cy = c[1], cz = c[2];
    int cnt = 0;
    for (int j = 0; j < N && cnt < K_NBR; j++) {
        float dx = pb[j * 3] - cx, dy = pb[j * 3 + 1] - cy, dz = pb[j * 3 + 2] - cz;
        float d2 = dx * dx + dy * dy + dz * dz;
        if (d2 < r2) { nb[cnt++] = j; }
    }
    int first = (cnt > 0) ? nb[0] : 0;
    for (int q = cnt; q < K_NBR; q++) nb[q] = first;
}

// ----------------------------------------------------------------------------
// Stage-1 GEMM input x0 [S*K, 32] f16 (rows 0-2 dp, 3-5 df, 6-31 zero).
// ----------------------------------------------------------------------------
__global__ void build_x1_kernel(const float* __restrict__ pts, const float* __restrict__ ctr,
                                const float* __restrict__ f,   const float* __restrict__ cf,
                                const int* __restrict__ nbr, _Float16* __restrict__ x,
                                int S, int N) {
    int Nt = S * K_NBR;
    int n = blockIdx.x * blockDim.x + threadIdx.x;
    if (n >= Nt) return;
    int s = n / K_NBR;
    int j = nbr[n];
    _Float16* xr = x + (size_t)n * 32;
#pragma unroll
    for (int d = 0; d < 3; d++)
        xr[d] = (_Float16)(pts[j * 3 + d] - ctr[s * 3 + d]);
#pragma unroll
    for (int c = 0; c < 3; c++)
        xr[3 + c] = (_Float16)(f[(size_t)c * N + j] - cf[(size_t)c * S + s]);
#pragma unroll
    for (int r = 6; r < 32; r++) xr[r] = (_Float16)0.0f;
}

// Stage-2 GEMM input x0 [S*K, 224] f16 (0-2 dp, 3-194 df, 195-223 zero).
// f (N,C) row-major f16; cf (S,C) f16.
__global__ void build_x2_kernel(const float* __restrict__ pts, const float* __restrict__ ctr,
                                const _Float16* __restrict__ f, const _Float16* __restrict__ cf,
                                const int* __restrict__ nbr, _Float16* __restrict__ x,
                                int S, int N, int C) {
    int Nt = S * K_NBR;
    int n = blockIdx.x * blockDim.x + threadIdx.x;
    if (n >= Nt) return;
    int s = n / K_NBR;
    int j = nbr[n];
    _Float16* xr = x + (size_t)n * 224;
#pragma unroll
    for (int d = 0; d < 3; d++)
        xr[d] = (_Float16)(pts[j * 3 + d] - ctr[s * 3 + d]);
    const _Float16* fr  = f  + (size_t)j * C;
    const _Float16* cfr = cf + (size_t)s * C;
    for (int c = 0; c < C; c++)
        xr[3 + c] = (_Float16)((float)fr[c] - (float)cfr[c]);
#pragma unroll
    for (int r = 195; r < 224; r++) xr[r] = (_Float16)0.0f;
}

// ----------------------------------------------------------------------------
// Weight convert + K-pad: wh[o, k] = k < Cin ? (f16)w[o*Cin+k] : 0, k < Kpad.
// ----------------------------------------------------------------------------
__global__ void cvt_w_kernel(const float* __restrict__ w, _Float16* __restrict__ wh,
                             int Cout, int Cin, int Kpad) {
    int t = blockIdx.x * blockDim.x + threadIdx.x;
    if (t >= Cout * Kpad) return;
    int o = t / Kpad, k = t % Kpad;
    wh[t] = (k < Cin) ? (_Float16)w[(size_t)o * Cin + k] : (_Float16)0.0f;
}

// ----------------------------------------------------------------------------
// WMMA GEMM:  Y[n, m] = sum_k W[m, k] * X[n, k] + bias[m]
//   Wh   : f16 [Cout, Ktot]  (K-padded weights)
//   X0   : f16 [S, c0]       pooled branch, row = n>>5, covers k in [0, c0)
//   X1   : f16 [Nt, ldx]     covers k in [c0, Ktot), ldx = Ktot - c0
//   Y    : f16 [Nt, Cout]
// One wave = 16 output channels x 64 columns (4 WMMA tiles, A reused).
// All fragment loads are 16-byte b128s; no bounds checks (everything padded).
// ----------------------------------------------------------------------------
__global__ void __launch_bounds__(256)
gemm_wmma_kernel(const _Float16* __restrict__ Wh, const float* __restrict__ bias,
                 const _Float16* X0, int c0,
                 const _Float16* X1, int ldx,
                 _Float16* __restrict__ Y,
                 int Cout, int Ktot, int Nt) {
    int wave = (int)((blockIdx.x * blockDim.x + threadIdx.x) >> 5);
    int lane = (int)(threadIdx.x & 31);
    int tilesN = Nt >> 6;                       // 64 columns per wave
    int tileM = wave / tilesN;
    int tileN = wave % tilesN;
    if (tileM * 16 >= Cout) return;

    int m0 = tileM * 16;
    int n0 = tileN * 64;
    int row = lane & 15, half = lane >> 4;

    const _Float16* wrow = Wh + (size_t)(m0 + row) * Ktot;
    const _Float16* xrow[4];
    const _Float16* prow[4];
#pragma unroll
    for (int j = 0; j < 4; j++) {
        int coln = n0 + 16 * j + row;
        xrow[j] = X1 + (size_t)coln * ldx;
        prow[j] = X0 + (size_t)(coln >> 5) * c0;   // no deref when c0 == 0
    }

    auto ldB = [&](const _Float16* base) -> v16h {
        v8h l0 = *(const v8h*)(base + 8 * half);
        v8h l1 = *(const v8h*)(base + 16 + 8 * half);
        return __builtin_shufflevector(l0, l1, 0, 1, 2, 3, 4, 5, 6, 7,
                                       8, 9, 10, 11, 12, 13, 14, 15);
    };
    auto ldA = [&](const _Float16* base) -> v16h {
        v8h l0 = *(const v8h*)(base + 8 * half);
        v8h l1 = *(const v8h*)(base + 16 + 8 * half);
        return __builtin_shufflevector(l0, l1, 0, 1, 2, 3, 4, 5, 6, 7,
                                       8, 9, 10, 11, 12, 13, 14, 15);
    };

    v8f acc[4] = {v8f{}, v8f{}, v8f{}, v8f{}};

    for (int kk = 0; kk < c0; kk += 32) {          // pooled-broadcast rows
        v16h af = ldA(wrow + kk);
        v16h bf[4];
#pragma unroll
        for (int j = 0; j < 4; j++) bf[j] = ldB(prow[j] + kk);
#pragma unroll
        for (int j = 0; j < 4; j++)
            acc[j] = __builtin_amdgcn_wmma_f32_16x16x32_f16(
                false, af, false, bf[j], (short)0, acc[j], false, false);
    }
    for (int kk = c0; kk < Ktot; kk += 32) {       // dense rows
        v16h af = ldA(wrow + kk);
        v16h bf[4];
#pragma unroll
        for (int j = 0; j < 4; j++) bf[j] = ldB(xrow[j] + (kk - c0));
#pragma unroll
        for (int j = 0; j < 4; j++)
            acc[j] = __builtin_amdgcn_wmma_f32_16x16x32_f16(
                false, af, false, bf[j], (short)0, acc[j], false, false);
    }

    float bv[8];
#pragma unroll
    for (int r = 0; r < 8; r++) bv[r] = bias[m0 + 8 * half + r];

#pragma unroll
    for (int j = 0; j < 4; j++) {
        int coln = n0 + 16 * j + row;
        v8h o;
#pragma unroll
        for (int r = 0; r < 8; r++) o[r] = (_Float16)(acc[j][r] + bv[r]);
        *(v8h*)(Y + (size_t)coln * Cout + m0 + 8 * half) = o;
    }
}

// ----------------------------------------------------------------------------
// InstanceNorm over [Nt, C] layout, deterministic 3-kernel reduction.
// ----------------------------------------------------------------------------
#define IN_NBLK 256

// K1: per-block partial sums, coalesced reads, register accumulators.
__global__ void in_part_kernel(const _Float16* __restrict__ X, int Nt, int C,
                               float* __restrict__ part) {
    int blk = blockIdx.x, tid = threadIdx.x;
    int rows = Nt / IN_NBLK;
    int nbeg = blk * rows;
    float s0 = 0, s1 = 0, s2 = 0, q0 = 0, q1 = 0, q2 = 0;
    for (int n = nbeg; n < nbeg + rows; n++) {
        const _Float16* rowp = X + (size_t)n * C;
        int c = tid;
        if (c < C) { float v = (float)rowp[c]; s0 += v; q0 += v * v; }
        c += 256;
        if (c < C) { float v = (float)rowp[c]; s1 += v; q1 += v * v; }
        c += 256;
        if (c < C) { float v = (float)rowp[c]; s2 += v; q2 += v * v; }
    }
    int c = tid;
    if (c < C) { part[(size_t)blk * C + c] = s0; part[(size_t)(IN_NBLK + blk) * C + c] = q0; }
    c += 256;
    if (c < C) { part[(size_t)blk * C + c] = s1; part[(size_t)(IN_NBLK + blk) * C + c] = q1; }
    c += 256;
    if (c < C) { part[(size_t)blk * C + c] = s2; part[(size_t)(IN_NBLK + blk) * C + c] = q2; }
}

// K2: finalize mean / inv-std per channel.
__global__ void in_stats_kernel(const float* __restrict__ part, int C, float invN,
                                float eps, float* __restrict__ stats) {
    int c = blockIdx.x * blockDim.x + threadIdx.x;
    if (c >= C) return;
    float s = 0, q = 0;
    for (int b = 0; b < IN_NBLK; b++) {
        s += part[(size_t)b * C + c];
        q += part[(size_t)(IN_NBLK + b) * C + c];
    }
    float m = s * invN;
    float var = q * invN - m * m;
    stats[c] = m;
    stats[C + c] = rsqrtf(var + eps);
}

// K3: apply (x - m) * inv, ReLU, coalesced.
__global__ void in_apply_kernel(_Float16* __restrict__ X, int total, int C,
                                const float* __restrict__ stats) {
    int t = blockIdx.x * blockDim.x + threadIdx.x;
    if (t >= total) return;
    int c = t % C;
    float v = ((float)X[t] - stats[c]) * stats[C + c];
    X[t] = (_Float16)(v > 0.0f ? v : 0.0f);
}

// ----------------------------------------------------------------------------
// Max over K neighbors: X [S*K, C] -> pooled/outh [S, C] f16, outf [C, S] f32.
// ----------------------------------------------------------------------------
__global__ void maxpool_kernel(const _Float16* __restrict__ X, int C, int S,
                               _Float16* pooled, _Float16* outh, float* outf) {
    int t = blockIdx.x * blockDim.x + threadIdx.x;
    if (t >= S * C) return;
    int s = t / C, c = t % C;
    const _Float16* p = X + ((size_t)s * K_NBR) * C + c;
    float mx = -3.0e38f;
#pragma unroll
    for (int k = 0; k < K_NBR; k++) {
        float v = (float)p[(size_t)k * C];
        if (v > mx) mx = v;
    }
    if (pooled) pooled[t] = (_Float16)mx;
    if (outh)   outh[t]   = (_Float16)mx;
    if (outf)   outf[(size_t)c * S + s] = mx;
}

// ----------------------------------------------------------------------------
static inline void launch_gemm(hipStream_t st, const _Float16* Wh, const float* bias,
                               const _Float16* X0, int c0, const _Float16* X1, int ldx,
                               _Float16* Y, int Cout, int Ktot, int Nt) {
    int waves = (Cout / 16) * (Nt / 64);
    int blocks = (waves + 7) / 8;   // 8 waves (256 threads) per block
    gemm_wmma_kernel<<<blocks, 256, 0, st>>>(Wh, bias, X0, c0, X1, ldx, Y, Cout, Ktot, Nt);
}

static inline void launch_inorm(hipStream_t st, _Float16* X, int Nt, int C,
                                float* part, float* stats, float eps) {
    in_part_kernel<<<IN_NBLK, 256, 0, st>>>(X, Nt, C, part);
    in_stats_kernel<<<(C + 255) / 256, 256, 0, st>>>(part, C, 1.0f / (float)Nt, eps, stats);
    int total = Nt * C;
    in_apply_kernel<<<(total + 255) / 256, 256, 0, st>>>(X, total, C, stats);
}

extern "C" void kernel_launch(void* const* d_in, const int* in_sizes, int n_in,
                              void* d_out, int out_size, void* d_ws, size_t ws_size,
                              hipStream_t stream) {
    (void)in_sizes; (void)n_in; (void)out_size; (void)ws_size;

    const float* p = (const float*)d_in[0];
    const float* f = (const float*)d_in[1];
    const float* w11a = (const float*)d_in[2];  const float* b11a = (const float*)d_in[3];
    const float* w11b = (const float*)d_in[4];  const float* b11b = (const float*)d_in[5];
    const float* w12a = (const float*)d_in[6];  const float* b12a = (const float*)d_in[7];
    const float* w12b = (const float*)d_in[8];  const float* b12b = (const float*)d_in[9];
    const float* w21a = (const float*)d_in[10]; const float* b21a = (const float*)d_in[11];
    const float* w21b = (const float*)d_in[12]; const float* b21b = (const float*)d_in[13];
    const float* w22a = (const float*)d_in[14]; const float* b22a = (const float*)d_in[15];
    const float* w22b = (const float*)d_in[16]; const float* b22b = (const float*)d_in[17];

    const int B = 4, N1 = 8192, S1 = 2048, S2 = 512;
    const int Nt1 = S1 * K_NBR;   // 65536 columns per batch (stage 1)
    const int Nt2 = S2 * K_NBR;   // 16384 columns per batch (stage 2)
    const float R2 = 0.1f * 0.1f;
    const float EPS = 1e-5f;

    // Output layout: p1 (B,S1,3) | f1 (B,192,S1) | p2 (B,S2,3) | f2 (B,384,S2)
    float* out = (float*)d_out;
    float* p1  = out;
    float* f1o = out + (size_t)B * S1 * 3;
    float* p2  = f1o + (size_t)B * 192 * S1;
    float* f2o = p2  + (size_t)B * S2 * 3;

    // Workspace bump allocator
    char* ws = (char*)d_ws;
    size_t off = 0;
    auto alloc = [&](size_t bytes) -> char* {
        char* r = ws + off;
        off += (bytes + 255) & ~(size_t)255;
        return r;
    };
    int* idx1 = (int*)alloc((size_t)B * S1 * 4);
    int* idx2 = (int*)alloc((size_t)B * S2 * 4);
    int* nbr1 = (int*)alloc((size_t)B * S1 * K_NBR * 4);
    int* nbr2 = (int*)alloc((size_t)B * S2 * K_NBR * 4);
    float*    cf1 = (float*)alloc((size_t)B * 3 * S1 * 4);       // (B,3,S1)
    _Float16* cf2 = (_Float16*)alloc((size_t)B * S2 * 192 * 2);  // (B,S2,192)
    _Float16* f1h = (_Float16*)alloc((size_t)B * S1 * 192 * 2);  // (B,S1,192)
    _Float16* x0  = (_Float16*)alloc((size_t)Nt2 * 224 * 2);     // >= Nt1*32*2
    _Float16* pooled = (_Float16*)alloc((size_t)S1 * 192 * 2);   // >= S2*384*2
    // padded f16 weights
    _Float16* wh11a = (_Float16*)alloc((size_t)192 * 32 * 2);
    _Float16* wh11b = (_Float16*)alloc((size_t)192 * 192 * 2);
    _Float16* wh12a = (_Float16*)alloc((size_t)384 * 384 * 2);
    _Float16* wh12b = (_Float16*)alloc((size_t)192 * 384 * 2);
    _Float16* wh21a = (_Float16*)alloc((size_t)384 * 224 * 2);
    _Float16* wh21b = (_Float16*)alloc((size_t)384 * 384 * 2);
    _Float16* wh22a = (_Float16*)alloc((size_t)768 * 768 * 2);
    _Float16* wh22b = (_Float16*)alloc((size_t)384 * 768 * 2);
    // instance-norm scratch
    float* in_part  = (float*)alloc((size_t)2 * IN_NBLK * 768 * 4);
    float* in_stats = (float*)alloc((size_t)2 * 768 * 4);
    // activation buffers, [Nt, C] layout
    _Float16* bufA = (_Float16*)alloc((size_t)Nt1 * 192 * 2);    // also bufD / buf2A / buf2D
    _Float16* bufB = (_Float16*)alloc((size_t)Nt1 * 192 * 2);    // also buf2B
    _Float16* bufC = (_Float16*)alloc((size_t)Nt1 * 384 * 2);    // also buf2C

    // Weight conversion (once per launch; tiny)
    auto cvt = [&](const float* w, _Float16* wh, int Cout, int Cin, int Kpad) {
        int tot = Cout * Kpad;
        cvt_w_kernel<<<(tot + 255) / 256, 256, 0, stream>>>(w, wh, Cout, Cin, Kpad);
    };
    cvt(w11a, wh11a, 192, 6, 32);
    cvt(w11b, wh11b, 192, 192, 192);
    cvt(w12a, wh12a, 384, 384, 384);
    cvt(w12b, wh12b, 192, 384, 384);
    cvt(w21a, wh21a, 384, 195, 224);
    cvt(w21b, wh21b, 384, 384, 384);
    cvt(w22a, wh22a, 768, 768, 768);
    cvt(w22b, wh22b, 384, 768, 768);

    // ======================= Stage 1 =======================
    fps_kernel<<<B, 256, 0, stream>>>(p, N1, S1, idx1);
    gather_p_kernel<<<(B * S1 + 255) / 256, 256, 0, stream>>>(p, idx1, p1, B, S1, N1);
    gather_f32_kernel<<<(B * 3 * S1 + 255) / 256, 256, 0, stream>>>(f, idx1, cf1, B, 3, S1, N1);
    ball_query_kernel<<<(B * S1 + 255) / 256, 256, 0, stream>>>(p1, p, nbr1, B, S1, N1, R2);

    for (int b = 0; b < B; b++) {
        const float* pb   = p   + (size_t)b * N1 * 3;
        const float* fb   = f   + (size_t)b * 3 * N1;
        const float* ctrb = p1  + (size_t)b * S1 * 3;
        const float* cfb  = cf1 + (size_t)b * 3 * S1;
        const int*   nbb  = nbr1 + (size_t)b * S1 * K_NBR;

        build_x1_kernel<<<(Nt1 + 255) / 256, 256, 0, stream>>>(pb, ctrb, fb, cfb, nbb, x0, S1, N1);

        launch_gemm(stream, wh11a, b11a, x0, 0, x0, 32, bufA, 192, 32, Nt1);
        launch_inorm(stream, bufA, Nt1, 192, in_part, in_stats, EPS);
        launch_gemm(stream, wh11b, b11b, bufA, 0, bufA, 192, bufB, 192, 192, Nt1);
        maxpool_kernel<<<(S1 * 192 + 255) / 256, 256, 0, stream>>>(bufB, 192, S1, pooled, nullptr, nullptr);
        // conv2a: k in [0,192) from pooled (row n>>5), k in [192,384) from bufB
        launch_gemm(stream, wh12a, b12a, pooled, 192, bufB, 192, bufC, 384, 384, Nt1);
        launch_inorm(stream, bufC, Nt1, 384, in_part, in_stats, EPS);
        launch_gemm(stream, wh12b, b12b, bufC, 0, bufC, 384, bufA, 192, 384, Nt1);
        launch_inorm(stream, bufA, Nt1, 192, in_part, in_stats, EPS);
        maxpool_kernel<<<(S1 * 192 + 255) / 256, 256, 0, stream>>>(
            bufA, 192, S1, nullptr,
            f1h + (size_t)b * S1 * 192, f1o + (size_t)b * 192 * S1);
    }

    // ======================= Stage 2 =======================
    fps_kernel<<<B, 256, 0, stream>>>(p1, S1, S2, idx2);
    gather_p_kernel<<<(B * S2 + 255) / 256, 256, 0, stream>>>(p1, idx2, p2, B, S2, S1);
    gather_f16_kernel<<<(B * S2 * 192 + 255) / 256, 256, 0, stream>>>(f1h, idx2, cf2, B, 192, S2, S1);
    ball_query_kernel<<<(B * S2 + 255) / 256, 256, 0, stream>>>(p2, p1, nbr2, B, S2, S1, R2);

    for (int b = 0; b < B; b++) {
        const float*    pb   = p1  + (size_t)b * S1 * 3;
        const float*    ctrb = p2  + (size_t)b * S2 * 3;
        const _Float16* fb   = f1h + (size_t)b * S1 * 192;
        const _Float16* cfb  = cf2 + (size_t)b * S2 * 192;
        const int*      nbb  = nbr2 + (size_t)b * S2 * K_NBR;

        build_x2_kernel<<<(Nt2 + 255) / 256, 256, 0, stream>>>(pb, ctrb, fb, cfb, nbb, x0, S2, S1, 192);

        launch_gemm(stream, wh21a, b21a, x0, 0, x0, 224, bufA, 384, 224, Nt2);
        launch_inorm(stream, bufA, Nt2, 384, in_part, in_stats, EPS);
        launch_gemm(stream, wh21b, b21b, bufA, 0, bufA, 384, bufB, 384, 384, Nt2);
        maxpool_kernel<<<(S2 * 384 + 255) / 256, 256, 0, stream>>>(bufB, 384, S2, pooled, nullptr, nullptr);
        launch_gemm(stream, wh22a, b22a, pooled, 384, bufB, 384, bufC, 768, 768, Nt2);
        launch_inorm(stream, bufC, Nt2, 768, in_part, in_stats, EPS);
        launch_gemm(stream, wh22b, b22b, bufC, 0, bufC, 768, bufA, 384, 768, Nt2);
        launch_inorm(stream, bufA, Nt2, 384, in_part, in_stats, EPS);
        maxpool_kernel<<<(S2 * 384 + 255) / 256, 256, 0, stream>>>(
            bufA, 384, S2, nullptr, nullptr, f2o + (size_t)b * 384 * S2);
    }
}
